// DSENFeatureExtractor_34351148434014
// MI455X (gfx1250) — matrix operand order repaired and checked by reading, not compiled, not run on hardware
//
#include <hip/hip_runtime.h>
#include <stdint.h>

typedef __attribute__((ext_vector_type(16))) _Float16 v16h;
typedef __attribute__((ext_vector_type(8)))  float    v8f;

union Frag16 { v16h v; uint32_t u[8]; _Float16 h[16]; };

// ---------------------------------------------------------------------------
// f32 -> f16 elementwise convert
// ---------------------------------------------------------------------------
__global__ void cvt_f16_kernel(const float* __restrict__ src,
                               _Float16* __restrict__ dst, size_t n) {
  size_t gid = (size_t)blockIdx.x * blockDim.x + threadIdx.x;
  if (gid < n) dst[gid] = (_Float16)src[gid];
}

// Pack conv weights [30][K] f32 -> [32][Kpad] f16 (zero pad rows/K)
__global__ void pack_convw_kernel(const float* __restrict__ w,
                                  _Float16* __restrict__ dst, int K, int Kpad) {
  size_t gid = (size_t)blockIdx.x * blockDim.x + threadIdx.x;
  size_t total = (size_t)32 * Kpad;
  if (gid >= total) return;
  int k = (int)(gid % Kpad);
  int o = (int)(gid / Kpad);
  float val = (o < 30 && k < K) ? w[(size_t)o * K + k] : 0.f;
  dst[gid] = (_Float16)val;
}

// ---------------------------------------------------------------------------
// Implicit-GEMM conv1d via WMMA (all shapes compile-time).
// out[s, m, p] = sum_{c,t} xh[row(s,c), p+t] * W[m, c*Kt+t]
//   A = W  [32 x Kpad] f16 (rows 30,31 zero)
//   B = patches, k=(c,t), n=p  (pairs (k,k+1) stay inside one channel row)
// block = 256 thr = 8 waves; wave -> one 16-col N tile x two 16-row M tiles
// ---------------------------------------------------------------------------
template <int SP, int LF, int LS, int Lout, int Kt, int K, int Kpad>
__global__ __launch_bounds__(256) void conv_wmma_kernel(
    const _Float16* __restrict__ xh,
    const _Float16* __restrict__ Aw,
    float* __restrict__ out) {
  int s    = blockIdx.x;
  int wave = threadIdx.x >> 5;
  int lane = threadIdx.x & 31;
  int ntile = blockIdx.y * 8 + wave;
  int n0 = ntile * 16;
  if (n0 >= Lout) return;
  int laneHi = lane >> 4;
  int lr     = lane & 15;
  int b = s / SP, seg = s % SP;
  const _Float16* xbase = xh + (size_t)(b * 30) * LF + (size_t)seg * LS;
  int n = n0 + lr;
  bool nvalid = (n < Lout);

  const _Float16* arow0 = Aw + (size_t)lr * Kpad;
  const _Float16* arow1 = Aw + (size_t)(lr + 16) * Kpad;

  v8f c0 = {}; v8f c1 = {};
  constexpr int ksteps = Kpad >> 5;
  for (int kt = 0; kt < ksteps; ++kt) {
    Frag16 a0, a1, bf;
#pragma unroll
    for (int vv = 0; vv < 8; ++vv) {
      int ka = kt * 32 + (laneHi ? 16 : 0) + 2 * vv;
      a0.u[vv] = *(const uint32_t*)(arow0 + ka);   // 4B aligned (Kpad even)
      a1.u[vv] = *(const uint32_t*)(arow1 + ka);
      int kb = kt * 32 + (laneHi ? 16 : 0) + 2 * vv;
      _Float16 e0 = (_Float16)0.f, e1 = (_Float16)0.f;
      if (nvalid && kb < K) {                      // K==Kpad: folds away
        int cc = kb / Kt, tt = kb % Kt;            // Kt const: shift/magic-mul
        const _Float16* p = xbase + (size_t)cc * LF + n + tt;
        e0 = p[0]; e1 = p[1];                      // parity unknown -> 2x u16
      }
      bf.h[2 * vv]     = e0;
      bf.h[2 * vv + 1] = e1;
    }
    c0 = __builtin_amdgcn_wmma_f32_16x16x32_f16(false, a0.v, false, bf.v,
                                                (short)0, c0, false, false);
    c1 = __builtin_amdgcn_wmma_f32_16x16x32_f16(false, a1.v, false, bf.v,
                                                (short)0, c1, false, false);
  }
  if (!nvalid) return;
#pragma unroll
  for (int vv = 0; vv < 8; ++vv) {
    int m0 = vv + laneHi * 8;                      // 0..15, always < 30
    out[((size_t)s * 30 + m0) * Lout + n] = c0[vv];
    int m1 = m0 + 16;
    if (m1 < 30) out[((size_t)s * 30 + m1) * Lout + n] = c1[vv];
  }
}

// ---------------------------------------------------------------------------
// BN (with folded conv bias) + ELU + adaptive avg pool.
// mode 0: local, regroup [2304,30,100] -> loc[256,30,900]
// mode 1: global, write xc[b,c,128+i] (bc stride 256)
// ---------------------------------------------------------------------------
__global__ void bnelu_pool_kernel(const float* __restrict__ conv,
                                  float* __restrict__ outp,
                                  const float* __restrict__ cb,
                                  const float* __restrict__ g,
                                  const float* __restrict__ be,
                                  const float* __restrict__ m,
                                  const float* __restrict__ v,
                                  int S, int Lin, int Lp, int mode) {
  size_t gid = (size_t)blockIdx.x * blockDim.x + threadIdx.x;
  size_t total = (size_t)S * 30 * Lp;
  if (gid >= total) return;
  int i = (int)(gid % Lp);
  size_t sc = gid / Lp;
  int c = (int)(sc % 30);
  int s = (int)(sc / 30);
  int st = (int)(((long long)i * Lin) / Lp);
  int en = (int)((((long long)(i + 1)) * Lin + Lp - 1) / Lp);
  float scale = g[c] * rsqrtf(v[c] + 1e-5f);
  float mm = m[c] - cb[c];
  float bb = be[c];
  float acc = 0.f;
  const float* row = conv + ((size_t)s * 30 + c) * Lin;
  for (int t = st; t < en; ++t) {
    float x = (row[t] - mm) * scale + bb;
    x = (x > 0.f) ? x : (expf(x) - 1.f);
    acc += x;
  }
  acc /= (float)(en - st);
  if (mode == 0) {
    int bq = s / 9, seg = s % 9;
    outp[((size_t)bq * 30 + c) * 900 + seg * 100 + i] = acc;
  } else {
    outp[((size_t)s * 30 + c) * 256 + 128 + i] = acc;
  }
}

// loc [256,30,900] -> xc[:, :, :128] (adaptive pool 900->128)
__global__ void pool2_kernel(const float* __restrict__ loc,
                             float* __restrict__ xc) {
  size_t gid = (size_t)blockIdx.x * blockDim.x + threadIdx.x;
  size_t total = (size_t)256 * 30 * 128;
  if (gid >= total) return;
  int i = (int)(gid % 128);
  size_t bc = gid / 128;
  int st = (i * 900) / 128;
  int en = ((i + 1) * 900 + 127) / 128;
  float acc = 0.f;
  const float* row = loc + bc * 900;
  for (int t = st; t < en; ++t) acc += row[t];
  xc[bc * 256 + i] = acc / (float)(en - st);
}

// ---------------------------------------------------------------------------
// EdgeConv projections: Ap[r,h] = X[r,:]@w1[h,:F] + b1[h], NB[r,h] = X[r,:]@w1[h,F:]
// rows = 7680 (B*30)
// ---------------------------------------------------------------------------
template <int F, int H>
__global__ __launch_bounds__(256) void anb_kernel(
    const float* __restrict__ X,
    const float* __restrict__ w1,
    const float* __restrict__ b1,
    float* __restrict__ Ap, float* __restrict__ NB) {
  size_t gid = (size_t)blockIdx.x * blockDim.x + threadIdx.x;
  size_t total = (size_t)7680 * H;
  if (gid >= total) return;
  int h = (int)(gid % H);
  size_t row = gid / H;
  const float* xr = X + row * F;
  const float* wa = w1 + (size_t)h * 2 * F;
  const float* wb = wa + F;
  float sa = 0.f, sb = 0.f;
#pragma unroll 4
  for (int f = 0; f < F; ++f) {
    float xv = xr[f];
    sa += xv * wa[f];
    sb += xv * wb[f];
  }
  Ap[gid] = sa + b1[h];
  NB[gid] = sb;
}

// ---------------------------------------------------------------------------
// EdgeConv main: per (b,i) block: E[j,h]=relu(Ap[b,i,h]+NB[b,j,h]) (f16, LDS),
// C[32 x O] = E @ w2h^T via WMMA, fused max over j (rows 0..29), +b2, write h.
// ---------------------------------------------------------------------------
template <int H, int O>
__global__ __launch_bounds__(256) void edge_wmma_kernel(
    const float* __restrict__ Ap,
    const float* __restrict__ NB,
    const _Float16* __restrict__ w2h,  // [O][H]
    const float* __restrict__ b2,
    float* __restrict__ hout) {
  __shared__ _Float16 E[32 * H];                 // [32][H]
  int bi = blockIdx.x;                           // b*30 + i
  int b  = bi / 30;
  const float* aprow  = Ap + (size_t)bi * H;
  const float* nbbase = NB + (size_t)b * 30 * H;
  constexpr int tot = 32 * H;
  for (int idx = threadIdx.x; idx < tot; idx += 256) {
    int j = idx / H, h = idx % H;
    float e = 0.f;
    if (j < 30) {
      float t = aprow[h] + nbbase[(size_t)j * H + h];
      e = (t > 0.f) ? t : 0.f;
    }
    E[idx] = (_Float16)e;
  }
  __syncthreads();

  int wave = threadIdx.x >> 5;
  int lane = threadIdx.x & 31;
  int laneHi = lane >> 4;
  int lr = lane & 15;
  constexpr int ksteps = H >> 5;
  constexpr int ntiles = O >> 4;
  const _Float16* Er0 = E + (size_t)lr * H;
  const _Float16* Er1 = E + (size_t)(lr + 16) * H;

#pragma unroll 1
  for (int nt = wave; nt < ntiles; nt += 8) {
    int n = nt * 16 + lr;
    const _Float16* brow = w2h + (size_t)n * H;
    v8f c0 = {}; v8f c1 = {};
#pragma unroll
    for (int kt = 0; kt < ksteps; ++kt) {
      Frag16 a0, a1, bf;
#pragma unroll
      for (int vv = 0; vv < 8; ++vv) {
        int ka = kt * 32 + (laneHi ? 16 : 0) + 2 * vv;
        a0.u[vv] = *(const uint32_t*)(Er0 + ka);       // LDS, 4B aligned
        a1.u[vv] = *(const uint32_t*)(Er1 + ka);
        int kb = kt * 32 + (laneHi ? 16 : 0) + 2 * vv;
        bf.u[vv] = *(const uint32_t*)(brow + kb);      // global, 4B aligned
      }
      c0 = __builtin_amdgcn_wmma_f32_16x16x32_f16(false, a0.v, false, bf.v,
                                                  (short)0, c0, false, false);
      c1 = __builtin_amdgcn_wmma_f32_16x16x32_f16(false, a1.v, false, bf.v,
                                                  (short)0, c1, false, false);
    }
    // max over j: c0 rows m=v+8*laneHi (0..15); c1 rows +16; rows 30,31 padded
    float mx = c0[0];
#pragma unroll
    for (int vv = 1; vv < 8; ++vv) mx = fmaxf(mx, c0[vv]);
#pragma unroll
    for (int vv = 0; vv < 8; ++vv) {
      int m1 = 16 + laneHi * 8 + vv;
      if (m1 < 30) mx = fmaxf(mx, c1[vv]);
    }
    mx = fmaxf(mx, __shfl_xor(mx, 16, 32));            // merge lane<->lane+16
    if (laneHi == 0)
      hout[(size_t)bi * O + nt * 16 + lr] = mx + b2[nt * 16 + lr];
  }
}

// ---------------------------------------------------------------------------
// Head: max over i of h1/h2/h3 -> feat[896] -> fc1 -> fc2. One block per b.
// ---------------------------------------------------------------------------
__global__ void head_kernel(const float* __restrict__ h1,
                            const float* __restrict__ h2,
                            const float* __restrict__ h3,
                            const float* __restrict__ fc1w,
                            const float* __restrict__ fc1b,
                            const float* __restrict__ fc2w,
                            const float* __restrict__ fc2b,
                            float* __restrict__ out) {
  __shared__ float feat[896];
  __shared__ float y1[256];
  int b = blockIdx.x;
  for (int f = threadIdx.x; f < 896; f += blockDim.x) {
    const float* src; int O, off;
    if (f < 128)      { src = h1 + (size_t)b * 30 * 128; O = 128; off = f; }
    else if (f < 384) { src = h2 + (size_t)b * 30 * 256; O = 256; off = f - 128; }
    else              { src = h3 + (size_t)b * 30 * 512; O = 512; off = f - 384; }
    float mx = src[off];
    for (int i = 1; i < 30; ++i) mx = fmaxf(mx, src[(size_t)i * O + off]);
    feat[f] = mx;
  }
  __syncthreads();
  for (int o = threadIdx.x; o < 256; o += blockDim.x) {
    float s = fc1b[o];
    const float* w = fc1w + (size_t)o * 896;
    for (int k = 0; k < 896; ++k) s += feat[k] * w[k];
    y1[o] = s;
  }
  __syncthreads();
  for (int o = threadIdx.x; o < 128; o += blockDim.x) {
    float s = fc2b[o];
    const float* w = fc2w + (size_t)o * 256;
    for (int k = 0; k < 256; ++k) s += y1[k] * w[k];
    out[(size_t)b * 128 + o] = s;
  }
}

// ---------------------------------------------------------------------------
static inline dim3 blocks1d(size_t n) { return dim3((unsigned)((n + 255) / 256)); }

extern "C" void kernel_launch(void* const* d_in, const int* in_sizes, int n_in,
                              void* d_out, int out_size, void* d_ws, size_t ws_size,
                              hipStream_t stream) {
  (void)in_sizes; (void)n_in; (void)out_size; (void)ws_size;
  const float* x       = (const float*)d_in[0];
  const float* convl_w = (const float*)d_in[1];
  const float* convl_b = (const float*)d_in[2];
  const float* bnl_g   = (const float*)d_in[3];
  const float* bnl_b   = (const float*)d_in[4];
  const float* bnl_m   = (const float*)d_in[5];
  const float* bnl_v   = (const float*)d_in[6];
  const float* convg_w = (const float*)d_in[7];
  const float* convg_b = (const float*)d_in[8];
  const float* bng_g   = (const float*)d_in[9];
  const float* bng_b   = (const float*)d_in[10];
  const float* bng_m   = (const float*)d_in[11];
  const float* bng_v   = (const float*)d_in[12];
  const float* ec1_w1  = (const float*)d_in[13];
  const float* ec1_w2  = (const float*)d_in[14];
  const float* ec2_w1  = (const float*)d_in[15];
  const float* ec2_w2  = (const float*)d_in[16];
  const float* ec3_w1  = (const float*)d_in[17];
  const float* ec3_w2  = (const float*)d_in[18];
  const float* fc1_w   = (const float*)d_in[19];
  const float* fc2_w   = (const float*)d_in[20];
  const float* ec1_b1  = (const float*)d_in[21];
  const float* ec1_b2  = (const float*)d_in[22];
  const float* ec2_b1  = (const float*)d_in[23];
  const float* ec2_b2  = (const float*)d_in[24];
  const float* ec3_b1  = (const float*)d_in[25];
  const float* ec3_b2  = (const float*)d_in[26];
  const float* fc1_b   = (const float*)d_in[27];
  const float* fc2_b   = (const float*)d_in[28];
  float* out = (float*)d_out;

  char* base = (char*)d_ws;
  size_t off = 0;
  auto take = [&](size_t bytes) -> void* {
    void* r = base + off;
    off += (bytes + 255) & ~(size_t)255;
    return r;
  };
  const size_t NXH = (size_t)256 * 30 * 3600;          // x in f16
  _Float16* xh   = (_Float16*)take(NXH * 2);
  _Float16* Awl  = (_Float16*)take((size_t)32 * 1920 * 2);
  _Float16* Awg  = (_Float16*)take((size_t)32 * 6016 * 2);
  _Float16* w2h1 = (_Float16*)take((size_t)128 * 128 * 2);
  _Float16* w2h2 = (_Float16*)take((size_t)256 * 256 * 2);
  _Float16* w2h3 = (_Float16*)take((size_t)512 * 512 * 2);
  float* convs = (float*)take((size_t)256 * 30 * 3401 * 4); // shared local/global conv scratch
  float* loc   = (float*)take((size_t)256 * 30 * 900 * 4);
  float* xc    = (float*)take((size_t)256 * 30 * 256 * 4);
  float* Ap    = (float*)take((size_t)7680 * 512 * 4);
  float* NBf   = (float*)take((size_t)7680 * 512 * 4);
  float* h1    = (float*)take((size_t)7680 * 128 * 4);
  float* h2    = (float*)take((size_t)7680 * 256 * 4);
  float* h3    = (float*)take((size_t)7680 * 512 * 4);

  // ---- precision conversion / weight packing ----
  cvt_f16_kernel<<<blocks1d(NXH), 256, 0, stream>>>(x, xh, NXH);
  pack_convw_kernel<<<blocks1d((size_t)32 * 1920), 256, 0, stream>>>(convl_w, Awl, 1920, 1920);
  pack_convw_kernel<<<blocks1d((size_t)32 * 6016), 256, 0, stream>>>(convg_w, Awg, 6000, 6016);
  cvt_f16_kernel<<<blocks1d((size_t)128 * 128), 256, 0, stream>>>(ec1_w2, w2h1, (size_t)128 * 128);
  cvt_f16_kernel<<<blocks1d((size_t)256 * 256), 256, 0, stream>>>(ec2_w2, w2h2, (size_t)256 * 256);
  cvt_f16_kernel<<<blocks1d((size_t)512 * 512), 256, 0, stream>>>(ec3_w2, w2h3, (size_t)512 * 512);

  // ---- local branch: conv (WMMA) -> BN+ELU+pool(337->100, regroup) -> pool(900->128) ----
  {
    dim3 grid(2304, (337 + 127) / 128);
    conv_wmma_kernel<9, 3600, 400, 337, 64, 1920, 1920>
        <<<grid, 256, 0, stream>>>(xh, Awl, convs);
  }
  bnelu_pool_kernel<<<blocks1d((size_t)2304 * 30 * 100), 256, 0, stream>>>(
      convs, loc, convl_b, bnl_g, bnl_b, bnl_m, bnl_v, 2304, 337, 100, 0);
  pool2_kernel<<<blocks1d((size_t)256 * 30 * 128), 256, 0, stream>>>(loc, xc);

  // ---- global branch: conv (WMMA) -> BN+ELU+pool(3401->128) into xc[:,:,128:] ----
  {
    dim3 grid(256, (3401 + 127) / 128);
    conv_wmma_kernel<1, 3600, 3600, 3401, 200, 6000, 6016>
        <<<grid, 256, 0, stream>>>(xh, Awg, convs);
  }
  bnelu_pool_kernel<<<blocks1d((size_t)256 * 30 * 128), 256, 0, stream>>>(
      convs, xc, convg_b, bng_g, bng_b, bng_m, bng_v, 256, 3401, 128, 1);

  // ---- EdgeConv 1: F=256, H=128, O=128 ----
  anb_kernel<256, 128><<<blocks1d((size_t)7680 * 128), 256, 0, stream>>>(
      xc, ec1_w1, ec1_b1, Ap, NBf);
  edge_wmma_kernel<128, 128><<<7680, 256, 0, stream>>>(Ap, NBf, w2h1, ec1_b2, h1);

  // ---- EdgeConv 2: F=128, H=256, O=256 ----
  anb_kernel<128, 256><<<blocks1d((size_t)7680 * 256), 256, 0, stream>>>(
      h1, ec2_w1, ec2_b1, Ap, NBf);
  edge_wmma_kernel<256, 256><<<7680, 256, 0, stream>>>(Ap, NBf, w2h2, ec2_b2, h2);

  // ---- EdgeConv 3: F=256, H=512, O=512 ----
  anb_kernel<256, 512><<<blocks1d((size_t)7680 * 512), 256, 0, stream>>>(
      h2, ec3_w1, ec3_b1, Ap, NBf);
  edge_wmma_kernel<512, 512><<<7680, 256, 0, stream>>>(Ap, NBf, w2h3, ec3_b2, h3);

  // ---- head: channel max + fc1 + fc2 ----
  head_kernel<<<256, 256, 0, stream>>>(h1, h2, h3, fc1_w, fc1_b, fc2_w, fc2_b, out);
}
